// Attention_5557687681160
// MI455X (gfx1250) — matrix operand order repaired and checked by reading, not compiled, not run on hardware
//
#include <hip/hip_runtime.h>

// ---------------------------------------------------------------------------
// MI455X (gfx1250) LoRA-GQA attention block, bf16 WMMA, wave32.
// ---------------------------------------------------------------------------

typedef __bf16 bf16_t;
typedef __bf16 v16bf __attribute__((ext_vector_type(16)));
typedef __bf16 v8bf  __attribute__((ext_vector_type(8)));
typedef float  v8f   __attribute__((ext_vector_type(8)));

#define B_    2
#define S_    1024
#define DIM_  4096
#define H_    32
#define KVH_  8
#define HD_   128
#define RANK_ 8
#define M_    (B_ * S_)          // 2048 total rows
#define KVD_  (KVH_ * HD_)       // 1024

static __device__ __forceinline__ v16bf cat8(v8bf lo, v8bf hi) {
  return __builtin_shufflevector(lo, hi, 0,1,2,3,4,5,6,7,8,9,10,11,12,13,14,15);
}

static __device__ __forceinline__ v8f wmma_bf16(v16bf a, v16bf b, v8f c) {
  // D = A(16x32 bf16) * B(32x16 bf16) + C(16x16 f32)
  return __builtin_amdgcn_wmma_f32_16x16x32_bf16(false, a, false, b,
                                                 (short)0, c, false, false);
}

// LDS byte-offset of a generic pointer to __shared__ storage (addrspacecast).
typedef __attribute__((address_space(3))) void lds_void_t;
static __device__ __forceinline__ unsigned lds_addr(const void* p) {
  return (unsigned)(unsigned long long)(lds_void_t*)p;
}

// GLOBAL_LOAD_ASYNC_TO_LDS_B128 in GVS mode:
//   LDS[lds_off + lane] = MEM[sbase + voff + lane]  (16B per lane, ASYNCcnt)
static __device__ __forceinline__ void async_ld_lds_b128(unsigned lds_off,
                                                         unsigned voff,
                                                         const void* sbase) {
  asm volatile("global_load_async_to_lds_b128 %0, %1, %2"
               :: "v"(lds_off), "v"(voff), "s"(sbase) : "memory");
}

static __device__ __forceinline__ void wait_asynccnt0() {
  asm volatile("s_wait_asynccnt 0x0" ::: "memory");
}

static __device__ __forceinline__ unsigned pack2(bf16_t a, bf16_t b) {
  unsigned short ua = __builtin_bit_cast(unsigned short, a);
  unsigned short ub = __builtin_bit_cast(unsigned short, b);
  return (unsigned)ua | ((unsigned)ub << 16);
}

// ---------------------------------------------------------------------------
// fp32 -> bf16 conversion (grid-stride, coalesced)
// ---------------------------------------------------------------------------
__global__ void cvt_f32_bf16(const float* __restrict__ in,
                             bf16_t* __restrict__ out, int n) {
  int i = blockIdx.x * blockDim.x + threadIdx.x;
  int stride = gridDim.x * blockDim.x;
  for (; i < n; i += stride) out[i] = (bf16_t)in[i];
}

// ---------------------------------------------------------------------------
// LoRA "down" projection: T[m][r] = sum_d X[m][d] * A[r][d]   (X bf16, A fp32)
// ---------------------------------------------------------------------------
__global__ __launch_bounds__(256) void lora_t_kernel(
    const bf16_t* __restrict__ X, const float* __restrict__ A,
    float* __restrict__ T) {
  const int w = threadIdx.x >> 5;
  const int lane = threadIdx.x & 31;
  const int row = blockIdx.x * 8 + w;
  const bf16_t* xp = X + (size_t)row * DIM_;
  float acc[RANK_] = {0.f, 0.f, 0.f, 0.f, 0.f, 0.f, 0.f, 0.f};
  for (int d = lane; d < DIM_; d += 32) {
    float xv = (float)xp[d];
#pragma unroll
    for (int r = 0; r < RANK_; r++) acc[r] += xv * A[r * DIM_ + d];
  }
#pragma unroll
  for (int r = 0; r < RANK_; r++) {
    float v = acc[r];
#pragma unroll
    for (int off = 16; off > 0; off >>= 1) v += __shfl_xor(v, off, 32);
    if (lane == 0) T[(size_t)row * RANK_ + r] = v;
  }
}

// ---------------------------------------------------------------------------
// bf16 WMMA GEMM:  Y[M x N] = X[M x 4096] * W[N x 4096]^T  + t * LB^T (LoRA)
// Block = 64(M) x 256(N), 8 waves (2x4), wave tile 32x64 (2x4 WMMA tiles).
// ---------------------------------------------------------------------------
constexpr int EPI_F32  = 0;
constexpr int EPI_BF16 = 1;
constexpr int EPI_ROPE = 2;

template <int EPI>
__global__ __launch_bounds__(256) void gemm_lora(
    const bf16_t* __restrict__ X, const bf16_t* __restrict__ W,
    const float* __restrict__ T, const float* __restrict__ LB,
    float* __restrict__ Yf, bf16_t* __restrict__ Yh,
    const float* __restrict__ fcos, const float* __restrict__ fsin,
    int Ntot) {
  const int lane = threadIdx.x & 31;
  const int w    = threadIdx.x >> 5;
  const int hf   = lane >> 4;    // which 16-lane half
  const int lr   = lane & 15;    // row (A) / col (B,C) index within 16
  const int wm = w >> 2, wn = w & 3;
  const int m_base = blockIdx.y * 64 + wm * 32;
  const int n_base = blockIdx.x * 256 + wn * 64;

  v8f acc[2][4] = {};

  for (int k0 = 0; k0 < DIM_; k0 += 32) {
    v16bf af[2];
    v16bf bfr[4];
#pragma unroll
    for (int i = 0; i < 2; i++) {
      // A-matrix 16x32 bf16: lanes 0-15 row m, K = hf*16..+15 (contiguous)
      const bf16_t* ap = X + (size_t)(m_base + i * 16 + lr) * DIM_ + k0 + hf * 16;
      af[i] = *(const v16bf*)ap;
      if (k0 + 256 < DIM_) __builtin_prefetch(ap + 256, 0, 1);
    }
#pragma unroll
    for (int j = 0; j < 4; j++) {
      // B-matrix 32x16: lane holds column n = lr, K = hf*16..+15 (contiguous)
      const bf16_t* bp = W + (size_t)(n_base + j * 16 + lr) * DIM_ + k0 + hf * 16;
      bfr[j] = *(const v16bf*)bp;
      if (k0 + 256 < DIM_) __builtin_prefetch(bp + 256, 0, 1);
    }
#pragma unroll
    for (int i = 0; i < 2; i++)
#pragma unroll
      for (int j = 0; j < 4; j++)
        acc[i][j] = wmma_bf16(af[i], bfr[j], acc[i][j]);
  }

  // Epilogue: fused rank-8 LoRA add, optional RoPE, store.
#pragma unroll
  for (int i = 0; i < 2; i++) {
#pragma unroll
    for (int j = 0; j < 4; j++) {
      const int col = n_base + j * 16 + lr;
      const float* lbp = LB + (size_t)col * RANK_;
      const float lb0 = lbp[0], lb1 = lbp[1], lb2 = lbp[2], lb3 = lbp[3];
      const float lb4 = lbp[4], lb5 = lbp[5], lb6 = lbp[6], lb7 = lbp[7];
#pragma unroll
      for (int r = 0; r < 8; r++) {
        const int row = m_base + i * 16 + hf * 8 + r;  // C/D row = hf*8 + vgpr
        const float* tp = T + (size_t)row * RANK_;
        float v = acc[i][j][r] + tp[0] * lb0 + tp[1] * lb1 + tp[2] * lb2 +
                  tp[3] * lb3 + tp[4] * lb4 + tp[5] * lb5 + tp[6] * lb6 +
                  tp[7] * lb7;
        if (EPI == EPI_F32) {
          Yf[(size_t)row * Ntot + col] = v;
        } else if (EPI == EPI_BF16) {
          Yh[(size_t)row * Ntot + col] = (bf16_t)v;
        } else {  // EPI_ROPE: pairs (2p, 2p+1) are adjacent lanes
          const int s = row & (S_ - 1);
          const int p = (col & (HD_ - 1)) >> 1;
          const float c  = fcos[s * (HD_ / 2) + p];
          const float sn = fsin[s * (HD_ / 2) + p];
          const float partner = __shfl_xor(v, 1, 32);
          const float o = (col & 1) ? (partner * sn + v * c)
                                    : (v * c - partner * sn);
          Yh[(size_t)row * Ntot + col] = (bf16_t)o;
        }
      }
    }
  }
}

// ---------------------------------------------------------------------------
// Flash attention (causal, GQA n_rep = 4). One WG = 128 q rows of one (b,h).
// 8 waves x 16 q rows. 32 keys / iteration: QK^T = 8 WMMAs, PV = 8 WMMAs.
// K tile staged once per WG via GLOBAL_LOAD_ASYNC_TO_LDS_B128 (ASYNCcnt);
// V staged transposed with packed b32 LDS stores; online softmax in regs.
// ---------------------------------------------------------------------------
__global__ __launch_bounds__(256) void attention_kernel(
    const bf16_t* __restrict__ Q, const bf16_t* __restrict__ K,
    const bf16_t* __restrict__ V, bf16_t* __restrict__ O) {
  __shared__ bf16_t ldsK[32][HD_];      // K tile: [key][hd]     (8 KB)
  __shared__ bf16_t ldsVT[HD_][32];     // V^T tile: [hd][key]   (8 KB)
  __shared__ bf16_t ldsP[8][16][32];    // per-wave P: [qrow][key] (8 KB)

  const int lane = threadIdx.x & 31;
  const int w = threadIdx.x >> 5;
  const int hf = lane >> 4;
  const int lr = lane & 15;
  const int bh = blockIdx.y;
  const int b = bh / H_, h = bh % H_;
  const int kvh = h / (H_ / KVH_);
  const int q0 = blockIdx.x * 128;
  const int mrow = q0 + w * 16;  // this wave's base query row

  // Preload Q as 4 A-frags (head dim 128 = 4 x 32)
  v16bf qf[4];
  {
    const bf16_t* qbase = Q + (size_t)(b * S_ + mrow + lr) * DIM_ + h * HD_;
#pragma unroll
    for (int kk = 0; kk < 4; kk++) {
      const bf16_t* qp = qbase + kk * 32 + hf * 16;
      qf[kk] = *(const v16bf*)qp;
    }
  }

  // per-thread staging coordinates
  const int skey = threadIdx.x >> 3;        // 0..31 : key row for K staging
  const int schk = threadIdx.x & 7;         // 0..7  : 16B chunk within row half
  const unsigned ldsK_base = lds_addr(&ldsK[0][0]);
  const int vkp = threadIdx.x >> 4;         // 0..15 : key pair for V staging
  const int vc  = threadIdx.x & 15;         // 0..15 : 8-elem hd chunk

  v8f o[8] = {};
  float mst[8], lst[8];
#pragma unroll
  for (int r = 0; r < 8; r++) { mst[r] = -1e30f; lst[r] = 0.f; }

  const int kend = q0 + 128;  // causal: keys beyond query tile never needed
  const float scale = 0.08838834764831845f;  // 1/sqrt(128)
  const float NEG_INF = -__builtin_inff();

  for (int k0 = 0; k0 < kend; k0 += 32) {
    __syncthreads();  // prior iteration's LDS readers are done

    // ---- stage K tile (32 keys x 128 hd) via async copy to LDS ----
    {
      const bf16_t* gbase = K + (size_t)(b * S_ + k0) * KVD_ + kvh * HD_;
      const unsigned voff = (unsigned)(skey * (KVD_ * 2) + schk * 16);
      const unsigned loff = ldsK_base + (unsigned)(skey * (HD_ * 2) + schk * 16);
      async_ld_lds_b128(loff, voff, gbase);
      async_ld_lds_b128(loff + 128u, voff + 128u, gbase);
    }

    // ---- stage V^T (packed 2 keys per dword, 8 x ds_store_b32) ----
    {
      const bf16_t* vp0 =
          V + (size_t)(b * S_ + k0 + 2 * vkp) * KVD_ + kvh * HD_ + vc * 8;
      v8bf va = *(const v8bf*)vp0;
      v8bf vb = *(const v8bf*)(vp0 + KVD_);
#pragma unroll
      for (int e = 0; e < 8; e++) {
        *(unsigned*)&ldsVT[vc * 8 + e][2 * vkp] = pack2(va[e], vb[e]);
      }
    }

    wait_asynccnt0();   // this wave's async K copies landed in LDS
    __syncthreads();    // all waves' K + V^T staging visible

    // ---- scores: S = Q * K^T for 32 keys (two 16-key N tiles) ----
    v8f s0 = {}, s1 = {};
#pragma unroll
    for (int kk = 0; kk < 4; kk++) {
      const bf16_t* kp0 = &ldsK[lr][kk * 32 + hf * 16];
      const bf16_t* kp1 = &ldsK[16 + lr][kk * 32 + hf * 16];
      s0 = wmma_bf16(qf[kk], *(const v16bf*)kp0, s0);
      s1 = wmma_bf16(qf[kk], *(const v16bf*)kp1, s1);
    }

    // ---- online softmax update (per row = hf*8 + r) ----
#pragma unroll
    for (int r = 0; r < 8; r++) {
      const int row = mrow + hf * 8 + r;
      const int key0 = k0 + lr, key1 = k0 + 16 + lr;
      float x0 = (key0 > row) ? NEG_INF : s0[r] * scale;
      float x1 = (key1 > row) ? NEG_INF : s1[r] * scale;
      float vmax = fmaxf(x0, x1);
#pragma unroll
      for (int off = 1; off < 16; off <<= 1)
        vmax = fmaxf(vmax, __shfl_xor(vmax, off, 32));
      const float mn = fmaxf(mst[r], vmax);
      const float alpha = __expf(mst[r] - mn);
      const float p0 = __expf(x0 - mn);
      const float p1 = __expf(x1 - mn);
      float rs = p0 + p1;
#pragma unroll
      for (int off = 1; off < 16; off <<= 1) rs += __shfl_xor(rs, off, 32);
      lst[r] = lst[r] * alpha + rs;
      mst[r] = mn;
#pragma unroll
      for (int t = 0; t < 8; t++) o[t][r] *= alpha;
      ldsP[w][hf * 8 + r][lr] = (bf16_t)p0;
      ldsP[w][hf * 8 + r][16 + lr] = (bf16_t)p1;
    }

    // ---- O += P * V  (P: 16x32 A-frag from LDS, V^T rows contiguous) ----
    v16bf pf;
    {
      const bf16_t* pp = &ldsP[w][lr][hf * 16];
      pf = *(const v16bf*)pp;
    }
#pragma unroll
    for (int t = 0; t < 8; t++) {
      const bf16_t* vtp = &ldsVT[t * 16 + lr][hf * 16];
      o[t] = wmma_bf16(pf, *(const v16bf*)vtp, o[t]);
    }
  }

  // normalize and store bf16 attention output in (B, S, H*HD) layout
#pragma unroll
  for (int t = 0; t < 8; t++) {
#pragma unroll
    for (int r = 0; r < 8; r++) {
      const int row = mrow + hf * 8 + r;
      const float val = o[t][r] / lst[r];
      O[(size_t)(b * S_ + row) * DIM_ + h * HD_ + t * 16 + lr] = (bf16_t)val;
    }
  }
}

// ---------------------------------------------------------------------------
// Host launcher
// ---------------------------------------------------------------------------
extern "C" void kernel_launch(void* const* d_in, const int* in_sizes, int n_in,
                              void* d_out, int out_size, void* d_ws,
                              size_t ws_size, hipStream_t stream) {
  (void)in_sizes; (void)n_in; (void)out_size; (void)ws_size;
  const float* x   = (const float*)d_in[0];
  const float* fc  = (const float*)d_in[1];
  const float* fs  = (const float*)d_in[2];
  // d_in[3] = mask (causality implemented directly)
  const float* wq  = (const float*)d_in[4];
  const float* wk  = (const float*)d_in[5];
  const float* wv  = (const float*)d_in[6];
  const float* wo  = (const float*)d_in[7];
  const float* lqa = (const float*)d_in[8];
  const float* lqb = (const float*)d_in[9];
  const float* lka = (const float*)d_in[10];
  const float* lkb = (const float*)d_in[11];
  const float* lva = (const float*)d_in[12];
  const float* lvb = (const float*)d_in[13];
  const float* loa = (const float*)d_in[14];
  const float* lob = (const float*)d_in[15];

  char* ws = (char*)d_ws;
  size_t off = 0;
  auto carve = [&](size_t bytes) -> char* {
    char* p = ws + off;
    off += (bytes + 255) & ~(size_t)255;
    return p;
  };
  bf16_t* xh  = (bf16_t*)carve((size_t)M_ * DIM_ * 2);
  bf16_t* wqh = (bf16_t*)carve((size_t)DIM_ * DIM_ * 2);
  bf16_t* wkh = (bf16_t*)carve((size_t)KVD_ * DIM_ * 2);
  bf16_t* wvh = (bf16_t*)carve((size_t)KVD_ * DIM_ * 2);
  bf16_t* woh = (bf16_t*)carve((size_t)DIM_ * DIM_ * 2);
  bf16_t* qh  = (bf16_t*)carve((size_t)M_ * DIM_ * 2);
  bf16_t* kh  = (bf16_t*)carve((size_t)M_ * KVD_ * 2);
  bf16_t* vh  = (bf16_t*)carve((size_t)M_ * KVD_ * 2);
  bf16_t* ah  = (bf16_t*)carve((size_t)M_ * DIM_ * 2);
  float*  tq  = (float*)carve((size_t)M_ * RANK_ * 4);
  float*  tk  = (float*)carve((size_t)M_ * RANK_ * 4);
  float*  tv  = (float*)carve((size_t)M_ * RANK_ * 4);
  float*  to  = (float*)carve((size_t)M_ * RANK_ * 4);

  // 1) fp32 -> bf16 conversions
  cvt_f32_bf16<<<4096, 256, 0, stream>>>(x,  xh,  M_ * DIM_);
  cvt_f32_bf16<<<4096, 256, 0, stream>>>(wq, wqh, DIM_ * DIM_);
  cvt_f32_bf16<<<2048, 256, 0, stream>>>(wk, wkh, KVD_ * DIM_);
  cvt_f32_bf16<<<2048, 256, 0, stream>>>(wv, wvh, KVD_ * DIM_);
  cvt_f32_bf16<<<4096, 256, 0, stream>>>(wo, woh, DIM_ * DIM_);

  // 2) LoRA down-projections t = x * A^T
  lora_t_kernel<<<M_ / 8, 256, 0, stream>>>(xh, lqa, tq);
  lora_t_kernel<<<M_ / 8, 256, 0, stream>>>(xh, lka, tk);
  lora_t_kernel<<<M_ / 8, 256, 0, stream>>>(xh, lva, tv);

  // 3) QKV projections (LoRA + RoPE fused in epilogue; bf16 outputs)
  dim3 blk(256);
  dim3 gq(DIM_ / 256, M_ / 64);
  dim3 gkv(KVD_ / 256, M_ / 64);
  gemm_lora<EPI_ROPE><<<gq,  blk, 0, stream>>>(xh, wqh, tq, lqb, nullptr, qh, fc, fs, DIM_);
  gemm_lora<EPI_ROPE><<<gkv, blk, 0, stream>>>(xh, wkh, tk, lkb, nullptr, kh, fc, fs, KVD_);
  gemm_lora<EPI_BF16><<<gkv, blk, 0, stream>>>(xh, wvh, tv, lvb, nullptr, vh, fc, fs, KVD_);

  // 4) flash attention (causal, GQA)
  attention_kernel<<<dim3(S_ / 128, B_ * H_), blk, 0, stream>>>(qh, kh, vh, ah);

  // 5) output projection with LoRA -> fp32 d_out
  lora_t_kernel<<<M_ / 8, 256, 0, stream>>>(ah, loa, to);
  gemm_lora<EPI_F32><<<gq, blk, 0, stream>>>(ah, woh, to, lob, (float*)d_out,
                                             nullptr, fc, fs, DIM_);
}
